// RNNSubLayer_23742579213119
// MI455X (gfx1250) — matrix operand order repaired and checked by reading, not compiled, not run on hardware
//
#include <hip/hip_runtime.h>
#include <hip/hip_bf16.h>

// ---------------------------------------------------------------------------
// RNN sub-layer on gfx1250 (MI455X), bf16 WMMA with fp32 accumulation.
//   input_seq: (64, 2048, 256) f32   hidden: (64, 256) f32
//   W: (256, 512) f32  [Wx | Wh]     b: (256,) f32
//   out: (64, 2048, 256) f32
//
// Phase 1: x_proj = input @ Wx^T + b   (memory-bound bf16 WMMA GEMM, in d_out)
// Phase 2: linear scan  h = xp + h @ Wh^T  via two-pass chunked scan:
//   partial (32 chunks x 64 steps, parallel) -> T^64 by squaring ->
//   combine (31 sequential chunk steps) -> final (parallel, writes d_out).
// Seed loads are software-pipelined one step ahead to hide L2 latency inside
// the latency-bound scan loops.
// ---------------------------------------------------------------------------

typedef __attribute__((ext_vector_type(16))) __bf16 v16bf;
typedef __attribute__((ext_vector_type(8)))  __bf16 v8bf;
typedef __attribute__((ext_vector_type(8)))  float  v8f;

#define INPUT_SIZE  256
#define HIDDEN_SIZE 256
#define SEQ_LEN     2048
#define BATCH       64
#define NROWS       (BATCH * SEQ_LEN)   // 131072 GEMM rows
#define CLEN        64                  // chunk length
#define CHUNKS      (SEQ_LEN / CLEN)    // 32
#define NSQUARE     6                   // T^(2^6) = T^64

// ---------------------------------------------------------------------------
// Pack W into WMMA-ready bf16 B-matrix fragments.
// Bpack[((nt*8 + ks)*32 + lane)*16 + j] = element K = ks*32+(lane>>4)*8+j (j<8),
// ks*32+16+(lane>>4)*8+(j-8) (j>=8), N = nt*16+(lane&15)  (one 32B load per fragment).
// ---------------------------------------------------------------------------
__global__ __launch_bounds__(256) void pack_w_kernel(const float* __restrict__ W,
                                                     __bf16* __restrict__ BpackX,
                                                     __bf16* __restrict__ BpackH) {
    int tid  = blockIdx.x * blockDim.x + threadIdx.x;   // 0..4095
    int lane = tid & 31;
    int ks   = (tid >> 5) & 7;
    int nt   = tid >> 8;
    int n    = nt * 16 + (lane & 15);
    int kbase = ks * 32 + (lane >> 4) * 8;

    v16bf vx, vh;
#pragma unroll
    for (int j = 0; j < 16; ++j) {
        int k = kbase + (j < 8 ? j : j + 8);
        vx[j] = (__bf16)W[n * 512 + k];                 // Wx^T[k][n]
        vh[j] = (__bf16)W[n * 512 + INPUT_SIZE + k];    // T[k][n] = Wh^T[k][n]
    }
    ((v16bf*)BpackX)[tid] = vx;
    ((v16bf*)BpackH)[tid] = vh;
}

// ---------------------------------------------------------------------------
// Phase 1: x_proj -> d_out. 8 waves/block; wave = one 16-row M-tile, all 16
// N-tiles; K=256 as 8 bf16 WMMA slabs. Half of BpackX (64 KB) is staged in
// LDS per block, cutting per-wave L2 B-fragment traffic roughly in half.
// ---------------------------------------------------------------------------
__global__ __launch_bounds__(256) void xproj_kernel(const float* __restrict__ in,
                                                    const __bf16* __restrict__ BpackX,
                                                    const float* __restrict__ bias,
                                                    float* __restrict__ out) {
    __shared__ __bf16 bsh[32768];   // 64 KB: fragments for nt = 0..7 (idx < 2048)

    int wave  = threadIdx.x >> 5;
    int lane  = threadIdx.x & 31;
    int half  = lane >> 4;
    int mrow  = lane & 15;
    int m0    = (blockIdx.x * 8 + wave) * 16;

    {   // cooperative stage of first half of BpackX (65536 B = 4096 x 16B)
        const float4* src = (const float4*)BpackX;
        float4* dst = (float4*)bsh;
        for (int i = threadIdx.x; i < 4096; i += 256) dst[i] = src[i];
    }

    const v16bf* bp  = (const v16bf*)BpackX;
    const v16bf* bls = (const v16bf*)bsh;

    v8f acc[16];
#pragma unroll
    for (int nt = 0; nt < 16; ++nt) {
        float bv = bias[nt * 16 + mrow];
#pragma unroll
        for (int r = 0; r < 8; ++r) acc[nt][r] = bv;
    }
    __syncthreads();

    const float* arow = in + (size_t)(m0 + mrow) * INPUT_SIZE;

#pragma unroll 1
    for (int ks = 0; ks < 8; ++ks) {
        int k0 = ks * 32;
        const float4* p = (const float4*)(arow + k0 + half * 8);
        float4 x0 = p[0], x1 = p[1];     // K+0..7
        float4 x2 = p[4], x3 = p[5];     // K+16..23
        v16bf a;
        a[0]=(__bf16)x0.x; a[1]=(__bf16)x0.y; a[2]=(__bf16)x0.z; a[3]=(__bf16)x0.w;
        a[4]=(__bf16)x1.x; a[5]=(__bf16)x1.y; a[6]=(__bf16)x1.z; a[7]=(__bf16)x1.w;
        a[8]=(__bf16)x2.x; a[9]=(__bf16)x2.y; a[10]=(__bf16)x2.z; a[11]=(__bf16)x2.w;
        a[12]=(__bf16)x3.x; a[13]=(__bf16)x3.y; a[14]=(__bf16)x3.z; a[15]=(__bf16)x3.w;

#pragma unroll
        for (int nt = 0; nt < 16; ++nt) {
            int fidx = (nt * 8 + ks) * 32 + lane;
            v16bf b = (nt < 8) ? bls[fidx] : bp[fidx];   // nt compile-time
            acc[nt] = __builtin_amdgcn_wmma_f32_16x16x32_bf16(
                false, a, false, b, (short)0, acc[nt], false, false);
        }
    }

#pragma unroll
    for (int nt = 0; nt < 16; ++nt) {
#pragma unroll
        for (int r = 0; r < 8; ++r) {
            int m = half * 8 + r;
            out[(size_t)(m0 + m) * HIDDEN_SIZE + nt * 16 + mrow] = acc[nt][r];
        }
    }
}

// ---------------------------------------------------------------------------
// Compute P = T^64 by 6 squarings. Single block (512 thr, 16 waves), P/Pt in
// global ws (L2-resident; __syncthreads orders block-level memory between
// phases). Emits BpackL = P in WMMA B-fragment layout.
// ---------------------------------------------------------------------------
__global__ __launch_bounds__(512) void power_kernel(const float* __restrict__ W,
                                                    __bf16* P, __bf16* Pt,
                                                    __bf16* __restrict__ BpackL) {
    int tid  = threadIdx.x;
    int wave = tid >> 5;
    int lane = tid & 31;
    int half = lane >> 4;
    int mrow = lane & 15;
    int m0   = wave * 16;

    // init P = T (T[r][c] = Wh[c][r] = W[c*512 + 256 + r]), Pt = T^T
    for (int i = tid; i < 256 * 256; i += 512) {
        int r = i >> 8, c = i & 255;
        __bf16 v = (__bf16)W[c * 512 + INPUT_SIZE + r];
        P[r * 256 + c]  = v;
        Pt[c * 256 + r] = v;
    }
    __syncthreads();

#pragma unroll 1
    for (int it = 0; it < NSQUARE; ++it) {
        v8f acc[16];
#pragma unroll
        for (int nt = 0; nt < 16; ++nt)
#pragma unroll
            for (int r = 0; r < 8; ++r) acc[nt][r] = 0.0f;

        const __bf16* prow = P + (m0 + mrow) * 256;
#pragma unroll 1
        for (int ks = 0; ks < 8; ++ks) {
            int k0 = ks * 32;
            v8bf lo = *(const v8bf*)(prow + k0 + half * 16);
            v8bf hi = *(const v8bf*)(prow + k0 + half * 16 + 8);
            v16bf a;
#pragma unroll
            for (int j = 0; j < 8; ++j) { a[j] = lo[j]; a[8 + j] = hi[j]; }
#pragma unroll
            for (int nt = 0; nt < 16; ++nt) {
                // B[k][n] = P[k][n]; contiguous from Pt row n
                v16bf b = *(const v16bf*)(Pt + (nt * 16 + mrow) * 256 + k0 + half * 16);
                acc[nt] = __builtin_amdgcn_wmma_f32_16x16x32_bf16(
                    false, a, false, b, (short)0, acc[nt], false, false);
            }
        }
        __syncthreads();   // all reads of P/Pt complete
#pragma unroll
        for (int nt = 0; nt < 16; ++nt)
#pragma unroll
            for (int r = 0; r < 8; ++r) {
                int m = m0 + half * 8 + r;
                int n = nt * 16 + mrow;
                __bf16 v = (__bf16)acc[nt][r];
                P[m * 256 + n]  = v;
                Pt[n * 256 + m] = v;
            }
        __syncthreads();   // P/Pt updated
    }

    // Emit BpackL (B-fragment layout of T^64) for the combine kernel.
    for (int idx = tid; idx < 16 * 8 * 32; idx += 512) {
        int l = idx & 31, ks = (idx >> 5) & 7, nt = idx >> 8;
        v16bf v;
#pragma unroll
        for (int j = 0; j < 16; ++j)
            v[j] = P[(ks * 32 + ((l >> 4) * 8) + (j < 8 ? j : j + 8)) * 256 + nt * 16 + (l & 15)];
        ((v16bf*)BpackL)[idx] = v;
    }
}

// ---------------------------------------------------------------------------
// Pass 1: per-chunk partial scan with zero initial state; store only the
// chunk-end state Pend[c] ([c][64][256] f32). 32 chunks x 4 batch tiles =
// 128 blocks; 16 waves = one N-tile each; Wh fragments in registers; the
// x_proj seed for step s+1 is prefetched during step s.
// ---------------------------------------------------------------------------
__global__ __launch_bounds__(512) void partial_kernel(const float* __restrict__ xp,
                                                      const __bf16* __restrict__ BpackH,
                                                      float* __restrict__ Pend) {
    __shared__ __bf16 hb[2][16][HIDDEN_SIZE];

    int wave = threadIdx.x >> 5;
    int lane = threadIdx.x & 31;
    int half = lane >> 4;
    int mrow = lane & 15;
    int chunk = blockIdx.x >> 2;
    int b0    = (blockIdx.x & 3) * 16;

    v16bf bslab[8];
    const v16bf* bp = (const v16bf*)BpackH;
#pragma unroll
    for (int ks = 0; ks < 8; ++ks) bslab[ks] = bp[(wave * 8 + ks) * 32 + lane];

    for (int i = threadIdx.x; i < 16 * HIDDEN_SIZE; i += blockDim.x)
        hb[0][i >> 8][i & 255] = (__bf16)0.0f;

    // prefetch seed for step 0
    v8f xpc;
#pragma unroll
    for (int r = 0; r < 8; ++r) {
        int m = half * 8 + r;
        xpc[r] = xp[((size_t)(b0 + m) * SEQ_LEN + chunk * CLEN) * HIDDEN_SIZE
                    + wave * 16 + mrow];
    }
    __syncthreads();

    int cur = 0;
#pragma unroll 1
    for (int ls = 0; ls < CLEN; ++ls) {
        int s = chunk * CLEN + ls;
        // prefetch next step's seed; latency hidden behind this step's chain
        v8f xpn;
        if (ls + 1 < CLEN) {
#pragma unroll
            for (int r = 0; r < 8; ++r) {
                int m = half * 8 + r;
                xpn[r] = xp[((size_t)(b0 + m) * SEQ_LEN + s + 1) * HIDDEN_SIZE
                            + wave * 16 + mrow];
            }
        }

        v8f acc = xpc;
        const __bf16* hr = &hb[cur][mrow][0];
#pragma unroll
        for (int ks = 0; ks < 8; ++ks) {
            int k0 = ks * 32;
            v8bf lo = *(const v8bf*)(hr + k0 + half * 8);
            v8bf hi = *(const v8bf*)(hr + k0 + 16 + half * 8);
            v16bf a;
#pragma unroll
            for (int j = 0; j < 8; ++j) { a[j] = lo[j]; a[8 + j] = hi[j]; }
            acc = __builtin_amdgcn_wmma_f32_16x16x32_bf16(
                false, a, false, bslab[ks], (short)0, acc, false, false);
        }

        int nxt = cur ^ 1;
#pragma unroll
        for (int r = 0; r < 8; ++r) {
            int m = half * 8 + r;
            hb[nxt][m][wave * 16 + mrow] = (__bf16)acc[r];
        }
        if (ls == CLEN - 1) {
#pragma unroll
            for (int r = 0; r < 8; ++r) {
                int m = half * 8 + r;
                Pend[(size_t)(chunk * BATCH + b0 + m) * HIDDEN_SIZE + wave * 16 + mrow] = acc[r];
            }
        }
        __syncthreads();
        cur = nxt;
        xpc = xpn;
    }
}

// ---------------------------------------------------------------------------
// Pass 2: sequential combine over chunks (31 steps, 4 blocks):
//   Hstart[0] = hidden ; Hstart[c] = Pend[c-1] + Hstart[c-1] @ T^64
// Pend seed prefetched one chunk ahead.
// ---------------------------------------------------------------------------
__global__ __launch_bounds__(512) void combine_kernel(const float* __restrict__ hidden,
                                                      const __bf16* __restrict__ BpackL,
                                                      const float* __restrict__ Pend,
                                                      float* __restrict__ Hstart) {
    __shared__ __bf16 hb[2][16][HIDDEN_SIZE];

    int wave = threadIdx.x >> 5;
    int lane = threadIdx.x & 31;
    int half = lane >> 4;
    int mrow = lane & 15;
    int b0   = blockIdx.x * 16;

    v16bf bslab[8];
    const v16bf* bp = (const v16bf*)BpackL;
#pragma unroll
    for (int ks = 0; ks < 8; ++ks) bslab[ks] = bp[(wave * 8 + ks) * 32 + lane];

    for (int i = threadIdx.x; i < 16 * HIDDEN_SIZE; i += blockDim.x) {
        int m = i >> 8, k = i & 255;
        float v = hidden[(size_t)(b0 + m) * HIDDEN_SIZE + k];
        hb[0][m][k] = (__bf16)v;
        Hstart[(size_t)(b0 + m) * HIDDEN_SIZE + k] = v;   // chunk 0
    }

    v8f xpc;   // seed for c=1: Pend[0]
#pragma unroll
    for (int r = 0; r < 8; ++r) {
        int m = half * 8 + r;
        xpc[r] = Pend[(size_t)(b0 + m) * HIDDEN_SIZE + wave * 16 + mrow];
    }
    __syncthreads();

    int cur = 0;
#pragma unroll 1
    for (int c = 1; c < CHUNKS; ++c) {
        v8f xpn;
        if (c + 1 < CHUNKS) {
#pragma unroll
            for (int r = 0; r < 8; ++r) {
                int m = half * 8 + r;
                xpn[r] = Pend[(size_t)(c * BATCH + b0 + m) * HIDDEN_SIZE + wave * 16 + mrow];
            }
        }

        v8f acc = xpc;
        const __bf16* hr = &hb[cur][mrow][0];
#pragma unroll
        for (int ks = 0; ks < 8; ++ks) {
            int k0 = ks * 32;
            v8bf lo = *(const v8bf*)(hr + k0 + half * 8);
            v8bf hi = *(const v8bf*)(hr + k0 + 16 + half * 8);
            v16bf a;
#pragma unroll
            for (int j = 0; j < 8; ++j) { a[j] = lo[j]; a[8 + j] = hi[j]; }
            acc = __builtin_amdgcn_wmma_f32_16x16x32_bf16(
                false, a, false, bslab[ks], (short)0, acc, false, false);
        }

        int nxt = cur ^ 1;
#pragma unroll
        for (int r = 0; r < 8; ++r) {
            int m = half * 8 + r;
            float v = acc[r];
            hb[nxt][m][wave * 16 + mrow] = (__bf16)v;
            Hstart[(size_t)(c * BATCH + b0 + m) * HIDDEN_SIZE + wave * 16 + mrow] = v;
        }
        __syncthreads();
        cur = nxt;
        xpc = xpn;
    }
}

// ---------------------------------------------------------------------------
// Pass 3: re-run each chunk seeded with its true Hstart[c]; write outputs in
// place over x_proj in d_out. 128 blocks in parallel; xp seed prefetched one
// step ahead (same-thread read-before-write keeps in-place update safe).
// ---------------------------------------------------------------------------
__global__ __launch_bounds__(512) void final_kernel(float* __restrict__ out,
                                                    const __bf16* __restrict__ BpackH,
                                                    const float* __restrict__ Hstart) {
    __shared__ __bf16 hb[2][16][HIDDEN_SIZE];

    int wave = threadIdx.x >> 5;
    int lane = threadIdx.x & 31;
    int half = lane >> 4;
    int mrow = lane & 15;
    int chunk = blockIdx.x >> 2;
    int b0    = (blockIdx.x & 3) * 16;

    v16bf bslab[8];
    const v16bf* bp = (const v16bf*)BpackH;
#pragma unroll
    for (int ks = 0; ks < 8; ++ks) bslab[ks] = bp[(wave * 8 + ks) * 32 + lane];

    for (int i = threadIdx.x; i < 16 * HIDDEN_SIZE; i += blockDim.x) {
        int m = i >> 8, k = i & 255;
        hb[0][m][k] = (__bf16)Hstart[(size_t)(chunk * BATCH + b0 + m) * HIDDEN_SIZE + k];
    }

    v8f xpc;
#pragma unroll
    for (int r = 0; r < 8; ++r) {
        int m = half * 8 + r;
        xpc[r] = out[((size_t)(b0 + m) * SEQ_LEN + chunk * CLEN) * HIDDEN_SIZE
                     + wave * 16 + mrow];
    }
    __syncthreads();

    int cur = 0;
#pragma unroll 1
    for (int ls = 0; ls < CLEN; ++ls) {
        int s = chunk * CLEN + ls;
        v8f xpn;
        if (ls + 1 < CLEN) {
#pragma unroll
            for (int r = 0; r < 8; ++r) {
                int m = half * 8 + r;
                xpn[r] = out[((size_t)(b0 + m) * SEQ_LEN + s + 1) * HIDDEN_SIZE
                             + wave * 16 + mrow];
            }
        }

        v8f acc = xpc;
        const __bf16* hr = &hb[cur][mrow][0];
#pragma unroll
        for (int ks = 0; ks < 8; ++ks) {
            int k0 = ks * 32;
            v8bf lo = *(const v8bf*)(hr + k0 + half * 8);
            v8bf hi = *(const v8bf*)(hr + k0 + 16 + half * 8);
            v16bf a;
#pragma unroll
            for (int j = 0; j < 8; ++j) { a[j] = lo[j]; a[8 + j] = hi[j]; }
            acc = __builtin_amdgcn_wmma_f32_16x16x32_bf16(
                false, a, false, bslab[ks], (short)0, acc, false, false);
        }

        int nxt = cur ^ 1;
#pragma unroll
        for (int r = 0; r < 8; ++r) {
            int m = half * 8 + r;
            float v = acc[r];
            hb[nxt][m][wave * 16 + mrow] = (__bf16)v;
            out[((size_t)(b0 + m) * SEQ_LEN + s) * HIDDEN_SIZE + wave * 16 + mrow] = v;
        }
        __syncthreads();
        cur = nxt;
        xpc = xpn;
    }
}

// ---------------------------------------------------------------------------
extern "C" void kernel_launch(void* const* d_in, const int* in_sizes, int n_in,
                              void* d_out, int out_size, void* d_ws, size_t ws_size,
                              hipStream_t stream) {
    const float* input_seq = (const float*)d_in[0];
    const float* hidden    = (const float*)d_in[1];
    const float* W         = (const float*)d_in[2];
    const float* bias      = (const float*)d_in[3];
    float*       out       = (float*)d_out;

    char* ws = (char*)d_ws;
    __bf16* BpackX = (__bf16*)(ws);                       // 128 KB
    __bf16* BpackH = (__bf16*)(ws + (128 << 10));         // 128 KB
    __bf16* BpackL = (__bf16*)(ws + (256 << 10));         // 128 KB
    __bf16* Pbuf   = (__bf16*)(ws + (384 << 10));         // 128 KB
    __bf16* Ptbuf  = (__bf16*)(ws + (512 << 10));         // 128 KB
    float*  Pend   = (float*)(ws + (640 << 10));          // 2 MB
    float*  Hstart = (float*)(ws + (640 << 10) + CHUNKS * BATCH * HIDDEN_SIZE * 4);

    pack_w_kernel<<<16, 256, 0, stream>>>(W, BpackX, BpackH);
    power_kernel<<<1, 512, 0, stream>>>(W, Pbuf, Ptbuf, BpackL);
    xproj_kernel<<<NROWS / (16 * 8), 256, 0, stream>>>(input_seq, BpackX, bias, out);
    partial_kernel<<<CHUNKS * 4, 512, 0, stream>>>(out, BpackH, Pend);
    combine_kernel<<<BATCH / 16, 512, 0, stream>>>(hidden, BpackL, Pend, Hstart);
    final_kernel<<<CHUNKS * 4, 512, 0, stream>>>(out, BpackH, Hstart);
}
